// MetaFuser_85383949845331
// MI455X (gfx1250) — matrix operand order repaired and checked by reading, not compiled, not run on hardware
//
#include <hip/hip_runtime.h>

#define D 256
#define NQ 4096
#define HEADS 8
#define DH 32

#define BM 128
#define BN 64
#define BK 32
#define LDSP 40     // padded LDS row stride in ushorts (80B = 5*16B, keeps uint4 alignment)
#define BPAD 8      // TDM pad: 4 DWORDs = 8 bf16 elems appended per B row
#define MAXK 256    // max K routed through gemm_bf16_k (conv has its own kernel)

typedef __attribute__((ext_vector_type(8)))  float   v8f;
typedef __attribute__((ext_vector_type(16))) __bf16  v16bf;

union frag16 {
  v16bf v;
  uint4 q[2];
  unsigned short u[16];
};

__device__ __forceinline__ unsigned short f32_to_bf16(float f) {
  unsigned int u = __float_as_uint(f);
  unsigned int r = u + 0x7FFFu + ((u >> 16) & 1u);          // round-to-nearest-even
  if ((u & 0x7FFFFFFFu) > 0x7F800000u) r = u | 0x00400000u; // quiet NaN
  return (unsigned short)(r >> 16);
}

// ---------------------------------------------------------------------------
// CDNA5 data-movement feature detection (compile-safe fallbacks everywhere)
// ---------------------------------------------------------------------------
#if defined(__has_builtin)
#if __has_builtin(__builtin_amdgcn_global_load_async_to_lds_b128)
#define HAVE_ASYNC_LDS 1
#endif
#if __has_builtin(__builtin_amdgcn_tensor_load_to_lds)
#define HAVE_TDM 1
#endif
#endif

#ifdef HAVE_ASYNC_LDS
#if __has_builtin(__builtin_amdgcn_s_wait_asynccnt)
#define WAIT_ASYNC(n) __builtin_amdgcn_s_wait_asynccnt(n)
#else
#define WAIT_ASYNC(n) asm volatile("s_wait_asynccnt %0" ::"i"(n) : "memory")
#endif
// builtin expects:  (int4 __device__* gsrc, int4 __shared__* ldst, i32 offset, i32 cpol)
typedef int v4i_gcc __attribute__((__vector_size__(16)));
__device__ __forceinline__ void async_copy16(void* lds, const void* gmem) {
  __builtin_amdgcn_global_load_async_to_lds_b128(
      (__attribute__((address_space(1))) v4i_gcc*)(gmem),
      (__attribute__((address_space(3))) v4i_gcc*)(lds), 0, 0);
}
#define STAGE16(dst, src) async_copy16((dst), (src))
#else
#define WAIT_ASYNC(n)
#define STAGE16(dst, src) (*(uint4*)(dst) = *(const uint4*)(src))
#endif

#ifdef HAVE_TDM
typedef __attribute__((ext_vector_type(4))) unsigned int u32x4;
typedef __attribute__((ext_vector_type(8))) int          i32x8;
typedef __attribute__((ext_vector_type(4))) int          i32x4;

__device__ __forceinline__ unsigned lds_offset_of(const void* p) {
  return (unsigned)(unsigned long long)(__attribute__((address_space(3))) const void*)p;
}

// 2D bf16 tile DMA: rows x rowElems from global (row stride strideElems) into
// LDS at ldsByte, LDS-padded by 4 DWORDs after every row (row = 2^(padCode+1) DW).
__device__ __forceinline__ void tdm_load_2d(const void* gtile, unsigned ldsByte,
                                            unsigned rowElems, unsigned rows,
                                            unsigned strideElems, unsigned padCode)
{
  const unsigned long long ga = (unsigned long long)gtile;
  u32x4 g0;
  g0.x = 1u;                                                   // count=1 (valid), user mode
  g0.y = ldsByte;                                              // lds_addr
  g0.z = (unsigned)ga;                                         // global_addr[31:0]
  g0.w = (unsigned)((ga >> 32) & 0x01FFFFFFull) | (2u << 30);  // global_addr[56:32] | type=2
  const unsigned w0 = (1u << 16)            // data_size = 2 bytes
                    | (1u << 20)            // pad_enable
                    | (padCode << 22)       // pad_interval (DWORDs per row)
                    | (3u << 25);           // pad_amount = 4 DWORDs
  const unsigned w1 = (rowElems & 0xFFFFu) << 16;                    // tensor_dim0[15:0]
  const unsigned w2 = (rowElems >> 16) | ((rows & 0xFFFFu) << 16);   // dim0 hi | dim1 lo
  const unsigned w3 = (rows >> 16) | ((rowElems & 0xFFFFu) << 16);   // dim1 hi | tile_dim0
  const unsigned w4 = rows & 0xFFFFu;                                // tile_dim1 (tile_dim2=0)
  const unsigned w5 = strideElems;                                   // tensor_dim0_stride lo32
  i32x8 g1 = { (int)w0, (int)w1, (int)w2, (int)w3, (int)w4, (int)w5, 0, 0 };
  i32x4 gz = { 0, 0, 0, 0 };
#if __clang_major__ >= 23
  i32x8 gz8 = { 0, 0, 0, 0, 0, 0, 0, 0 };
  __builtin_amdgcn_tensor_load_to_lds(g0, g1, gz, gz, gz8, 0);
#else
  __builtin_amdgcn_tensor_load_to_lds(g0, g1, gz, gz, 0);
#endif
}
#endif  // HAVE_TDM

// ---------------------------------------------------------------------------
// Generic bf16 WMMA GEMM:  out[M=4096, N] = A[M,K](bf16) * B[N,K](bf16)^T
//  - B panel [BN][K] staged ONCE per block via TDM (tensor_load_to_lds)
//  - A tiles double-buffered via global_load_async_to_lds_b128 (ASYNCcnt)
//  - epilogue: +bias, optional relu, optional residual, f32 and/or bf16 store
// ---------------------------------------------------------------------------
__global__ __launch_bounds__(256)
void gemm_bf16_k(const unsigned short* __restrict__ A, int lda,
                 const unsigned short* __restrict__ B, int ldb,
                 const float* __restrict__ bias,
                 const float* __restrict__ res,
                 float* __restrict__ outF, unsigned short* __restrict__ outH,
                 int ldo, int K, int relu, int padcode)
{
  __shared__ __align__(16) unsigned short As[2][BM * LDSP];
  __shared__ __align__(16) unsigned short Bs[BN * (MAXK + BPAD)];

  const int tid  = threadIdx.x;
  const int lane = tid & 31;
  const int wave = tid >> 5;
  const int m0   = blockIdx.x * BM;
  const int n0   = blockIdx.y * BN;
  const int bstr = K + BPAD;

  v8f acc[4];
#pragma unroll
  for (int j = 0; j < 4; ++j) acc[j] = v8f{0.f,0.f,0.f,0.f,0.f,0.f,0.f,0.f};

  const int arow   = tid >> 2;        // 0..63
  const int achunk = (tid & 3) * 8;   // 0,8,16,24 (ushorts)

  // ---- B panel staging (once per block) ----
#ifdef HAVE_TDM
  if (__builtin_amdgcn_readfirstlane((int)tid) < 32) {  // wave 0 only (scalar branch)
    tdm_load_2d((const char*)B + (size_t)n0 * ldb * 2, lds_offset_of(Bs),
                (unsigned)K, (unsigned)BN, (unsigned)ldb, (unsigned)padcode);
  }
#else
  (void)padcode;
  for (int e = tid * 8; e < BN * K; e += 256 * 8) {
    const int row = e / K, col = e % K;
    *(uint4*)&Bs[row * bstr + col] = *(const uint4*)&B[(size_t)(n0 + row) * ldb + col];
  }
#endif

  const int T = K / BK;
  // prologue: A tile 0 -> buffer 0
  STAGE16(&As[0][arow * LDSP + achunk],        &A[(size_t)(m0 + arow) * lda + achunk]);
  STAGE16(&As[0][(arow + 64) * LDSP + achunk], &A[(size_t)(m0 + arow + 64) * lda + achunk]);

  for (int t = 0; t < T; ++t) {
    const int k0 = t * BK;
    if (t + 1 < T) {
      const int kn = k0 + BK;
      STAGE16(&As[(t + 1) & 1][arow * LDSP + achunk],
              &A[(size_t)(m0 + arow) * lda + kn + achunk]);
      STAGE16(&As[(t + 1) & 1][(arow + 64) * LDSP + achunk],
              &A[(size_t)(m0 + arow + 64) * lda + kn + achunk]);
    }
    if (t + 2 < T)  // global_prefetch_b8 of the tile after next
      __builtin_prefetch(&A[(size_t)(m0 + arow) * lda + k0 + 2 * BK + achunk], 0, 3);
#ifdef HAVE_ASYNC_LDS
    if (t + 1 < T) { WAIT_ASYNC(2); } else { WAIT_ASYNC(0); }
#endif
#ifdef HAVE_TDM
    if (t == 0 && __builtin_amdgcn_readfirstlane((int)tid) < 32)
      __builtin_amdgcn_s_wait_tensorcnt(0);
#endif
    __syncthreads();  // tile t (A async + B TDM) visible to all waves

    const int r16 = lane & 15;
    const int ak  = (lane < 16) ? 0 : 16;  // A frag: K = ak..ak+15
    const int bk  = (lane < 16) ? 0 : 16;  // B frag: K = bk..bk+15
    frag16 fa;
    {
      const unsigned short* ap = &As[t & 1][(wave * 16 + r16) * LDSP];
      fa.q[0] = *(const uint4*)(ap + ak);
      fa.q[1] = *(const uint4*)(ap + ak + 8);
    }
#pragma unroll
    for (int j = 0; j < 4; ++j) {
      frag16 fb;
      const unsigned short* bp = &Bs[(j * 16 + r16) * bstr + k0 + bk];
      fb.q[0] = *(const uint4*)(bp);
      fb.q[1] = *(const uint4*)(bp + 8);
      acc[j] = __builtin_amdgcn_wmma_f32_16x16x32_bf16(
          false, fa.v, false, fb.v, (short)0, acc[j], false, false);
    }
    __syncthreads();  // all reads of buffer (t&1) done before it is re-staged
  }

  const int r16 = lane & 15;
  const int ro  = (lane < 16) ? 0 : 8;
#pragma unroll
  for (int j = 0; j < 4; ++j) {
    const int col = n0 + j * 16 + r16;
    const float bv = bias ? bias[col] : 0.f;
#pragma unroll
    for (int r = 0; r < 8; ++r) {
      const int row = m0 + wave * 16 + ro + r;
      float y = acc[j][r] + bv;
      if (relu) y = fmaxf(y, 0.f);
      if (res)  y += res[(size_t)row * ldo + col];
      if (outF) outF[(size_t)row * ldo + col] = y;
      if (outH) outH[(size_t)row * ldo + col] = f32_to_bf16(y);
    }
  }
}

// ---------------------------------------------------------------------------
// 3x3 conv (512 -> 256 over 64x64) as implicit GEMM, K = 9 taps * 512, ReLU.
// X: concat bf16 [4096][512] position-major, Wc: bf16 [256][4608] tap-major.
// Weight tiles staged via async-to-LDS; activation tiles boundary-masked.
// ---------------------------------------------------------------------------
__global__ __launch_bounds__(256)
void conv3x3_wmma_k(const unsigned short* __restrict__ X,
                    const unsigned short* __restrict__ Wc,
                    unsigned short* __restrict__ outH)
{
  __shared__ __align__(16) unsigned short As[BM * LDSP];
  __shared__ __align__(16) unsigned short Bs[BN * LDSP];

  const int tid  = threadIdx.x;
  const int lane = tid & 31;
  const int wave = tid >> 5;
  const int m0   = blockIdx.x * BM;
  const int n0   = blockIdx.y * BN;

  v8f acc[4];
#pragma unroll
  for (int j = 0; j < 4; ++j) acc[j] = v8f{0.f,0.f,0.f,0.f,0.f,0.f,0.f,0.f};

  const int arow   = tid >> 2;
  const int achunk = (tid & 3) * 8;

  for (int k0 = 0; k0 < 4608; k0 += BK) {
    STAGE16(&Bs[arow * LDSP + achunk],
            &Wc[(size_t)(n0 + arow) * 4608 + k0 + achunk]);
    const int k   = k0 + achunk;       // chunk never crosses a tap boundary
    const int tap = k / 512;
    const int c   = k - tap * 512;
    const int dy  = tap / 3 - 1;
    const int dx  = tap % 3 - 1;
#pragma unroll
    for (int half = 0; half < 2; ++half) {
      const int p  = m0 + arow + half * 64;
      const int yy = (p >> 6) + dy;
      const int xx = (p & 63) + dx;
      uint4 val = make_uint4(0u, 0u, 0u, 0u);
      if (yy >= 0 && yy < 64 && xx >= 0 && xx < 64)
        val = *(const uint4*)&X[(size_t)((yy << 6) + xx) * 512 + c];
      *(uint4*)&As[(arow + half * 64) * LDSP + achunk] = val;
    }
    if (k0 + BK < 4608)
      __builtin_prefetch(&Wc[(size_t)(n0 + arow) * 4608 + k0 + BK + achunk], 0, 3);
    WAIT_ASYNC(0);
    __syncthreads();

    const int r16 = lane & 15;
    const int ak  = (lane < 16) ? 0 : 16;
    const int bk  = (lane < 16) ? 0 : 16;
    frag16 fa;
    {
      const unsigned short* ap = &As[(wave * 16 + r16) * LDSP];
      fa.q[0] = *(const uint4*)(ap + ak);
      fa.q[1] = *(const uint4*)(ap + ak + 8);
    }
#pragma unroll
    for (int j = 0; j < 4; ++j) {
      frag16 fb;
      const unsigned short* bp = &Bs[(j * 16 + r16) * LDSP];
      fb.q[0] = *(const uint4*)(bp + bk);
      fb.q[1] = *(const uint4*)(bp + bk + 8);
      acc[j] = __builtin_amdgcn_wmma_f32_16x16x32_bf16(
          false, fa.v, false, fb.v, (short)0, acc[j], false, false);
    }
    __syncthreads();
  }

  const int r16 = lane & 15;
  const int ro  = (lane < 16) ? 0 : 8;
#pragma unroll
  for (int j = 0; j < 4; ++j) {
    const int col = n0 + j * 16 + r16;
#pragma unroll
    for (int r = 0; r < 8; ++r) {
      const int row = m0 + wave * 16 + ro + r;
      outH[(size_t)row * 256 + col] = f32_to_bf16(fmaxf(acc[j][r], 0.f));
    }
  }
}

// ---------------------------------------------------------------------------
// LayerNorm over D=256, one row per wave (8 rows / block).
// ---------------------------------------------------------------------------
__global__ __launch_bounds__(256)
void layernorm_k(const float* __restrict__ x, const float* __restrict__ s,
                 const float* __restrict__ b, const float* __restrict__ qpos,
                 unsigned short* __restrict__ t2h, unsigned short* __restrict__ t2qh)
{
  const int row  = blockIdx.x * 8 + (threadIdx.x >> 5);
  const int lane = threadIdx.x & 31;
  float v[8], sum = 0.f, sq = 0.f;
#pragma unroll
  for (int i = 0; i < 8; ++i) {
    v[i] = x[(size_t)row * 256 + i * 32 + lane];
    sum += v[i]; sq += v[i] * v[i];
  }
#pragma unroll
  for (int m = 16; m >= 1; m >>= 1) {
    sum += __shfl_xor(sum, m, 32);
    sq  += __shfl_xor(sq,  m, 32);
  }
  const float mean = sum * (1.f / 256.f);
  const float var  = sq * (1.f / 256.f) - mean * mean;
  const float inv  = rsqrtf(var + 1e-5f);
#pragma unroll
  for (int i = 0; i < 8; ++i) {
    const int c = i * 32 + lane;
    const float t = (v[i] - mean) * inv * s[c] + b[c];
    if (t2h)  t2h[(size_t)row * 256 + c]  = f32_to_bf16(t);
    if (t2qh) t2qh[(size_t)row * 256 + c] = f32_to_bf16(t + qpos[(size_t)row * 256 + c]);
  }
}

// ---------------------------------------------------------------------------
// MSDA sampling: one wave per (query, head); lane = dh channel.
// ---------------------------------------------------------------------------
__global__ __launch_bounds__(256)
void msda_k(const float* __restrict__ off, const float* __restrict__ awl,
            const float* __restrict__ val, unsigned short* __restrict__ outH)
{
  const int pair = blockIdx.x * 8 + (threadIdx.x >> 5);
  const int lane = threadIdx.x & 31;
  const int q = pair >> 3, h = pair & 7;

  const float* lg = &awl[(size_t)q * 64 + h * 8];
  float w[8], mx = -1e30f;
#pragma unroll
  for (int p = 0; p < 8; ++p) { w[p] = lg[p]; mx = fmaxf(mx, w[p]); }
  float den = 0.f;
#pragma unroll
  for (int p = 0; p < 8; ++p) { w[p] = __expf(w[p] - mx); den += w[p]; }
  const float invd = 1.f / den;

  const float qx = (float)(q & 63), qy = (float)(q >> 6);
  float acc = 0.f;
#pragma unroll
  for (int p = 0; p < 8; ++p) {
    const float ox = off[(size_t)q * 128 + h * 16 + p * 2 + 0];
    const float oy = off[(size_t)q * 128 + h * 16 + p * 2 + 1];
    const float xf = qx + ox, yf = qy + oy;
    const float xfl = floorf(xf), yfl = floorf(yf);
    const int x0 = (int)xfl, y0 = (int)yfl;
    const float wx = xf - xfl, wy = yf - yfl;
    float s = 0.f;
#pragma unroll
    for (int dy = 0; dy < 2; ++dy)
#pragma unroll
      for (int dx = 0; dx < 2; ++dx) {
        const int yy = y0 + dy, xx = x0 + dx;
        const float bw = (dy ? wy : 1.f - wy) * (dx ? wx : 1.f - wx);
        if (yy >= 0 && yy < 64 && xx >= 0 && xx < 64)
          s += bw * val[(size_t)((yy << 6) + xx) * 256 + h * 32 + lane];
      }
    acc += w[p] * invd * s;
  }
  outH[(size_t)q * 256 + h * 32 + lane] = f32_to_bf16(acc);
}

// ---------------------------------------------------------------------------
// Transpose V (f32 [key][h*32+d]) -> bf16 [h][d][key] for contiguous B frags.
// ---------------------------------------------------------------------------
__global__ __launch_bounds__(256)
void vtrans_k(const float* __restrict__ vf, unsigned short* __restrict__ vt)
{
  const int idx = blockIdx.x * 256 + threadIdx.x;  // over 4096*256
  const int key = idx >> 8, c = idx & 255;
  vt[(size_t)c * 4096 + key] = f32_to_bf16(vf[idx]);
}

// ---------------------------------------------------------------------------
// Flash attention: one wave per (head, 16-query tile); dh=32, 4096 keys.
// ---------------------------------------------------------------------------
__global__ __launch_bounds__(256)
void flash_k(const unsigned short* __restrict__ qk,
             const unsigned short* __restrict__ vt,
             unsigned short* __restrict__ outH)
{
  __shared__ __align__(16) unsigned short Pl[8][16 * LDSP];

  const int wave = threadIdx.x >> 5, lane = threadIdx.x & 31;
  unsigned short* P = &Pl[wave][0];
  const int gid = blockIdx.x * 8 + wave;
  const int h = gid >> 8, qt = gid & 255;
  const int q0 = qt * 16;
  const int r16 = lane & 15;
  const int ak  = (lane < 16) ? 0 : 16;
  const int bk  = (lane < 16) ? 0 : 16;

  frag16 fq;
  {
    const unsigned short* qp = &qk[(size_t)(q0 + r16) * 512 + h * 32];
    fq.q[0] = *(const uint4*)(qp + ak);
    fq.q[1] = *(const uint4*)(qp + ak + 8);
  }

  v8f o0 = v8f{0.f,0.f,0.f,0.f,0.f,0.f,0.f,0.f};
  v8f o1 = v8f{0.f,0.f,0.f,0.f,0.f,0.f,0.f,0.f};
  float m8[8], l8[8];
#pragma unroll
  for (int r = 0; r < 8; ++r) { m8[r] = -1e30f; l8[r] = 0.f; }
  const float scale = 0.17677669529663687f;  // 1/sqrt(32)
  const int mrow = (lane < 16) ? 0 : 8;

  for (int kk = 0; kk < 4096; kk += 32) {
    frag16 fk0, fk1;
    {
      const unsigned short* kp0 = &qk[(size_t)(kk + r16) * 512 + 256 + h * 32];
      fk0.q[0] = *(const uint4*)(kp0 + bk);
      fk0.q[1] = *(const uint4*)(kp0 + bk + 8);
      const unsigned short* kp1 = &qk[(size_t)(kk + 16 + r16) * 512 + 256 + h * 32];
      fk1.q[0] = *(const uint4*)(kp1 + bk);
      fk1.q[1] = *(const uint4*)(kp1 + bk + 8);
    }
    const v8f z = v8f{0.f,0.f,0.f,0.f,0.f,0.f,0.f,0.f};
    v8f s0 = __builtin_amdgcn_wmma_f32_16x16x32_bf16(false, fq.v, false, fk0.v, (short)0, z, false, false);
    v8f s1 = __builtin_amdgcn_wmma_f32_16x16x32_bf16(false, fq.v, false, fk1.v, (short)0, z, false, false);

    float p0[8], p1[8], alpha[8];
#pragma unroll
    for (int r = 0; r < 8; ++r) {
      const float sa = s0[r] * scale, sb = s1[r] * scale;
      float rm = fmaxf(sa, sb);
#pragma unroll
      for (int m = 8; m >= 1; m >>= 1) rm = fmaxf(rm, __shfl_xor(rm, m, 32));
      const float mn = fmaxf(m8[r], rm);
      const float al = __expf(m8[r] - mn);
      const float pa = __expf(sa - mn), pb = __expf(sb - mn);
      float rs = pa + pb;
#pragma unroll
      for (int m = 8; m >= 1; m >>= 1) rs += __shfl_xor(rs, m, 32);
      l8[r] = l8[r] * al + rs;
      m8[r] = mn;
      alpha[r] = al; p0[r] = pa; p1[r] = pb;
    }
#pragma unroll
    for (int r = 0; r < 8; ++r) { o0[r] *= alpha[r]; o1[r] *= alpha[r]; }

    // re-layout P (C layout) -> A fragment via per-wave LDS
#pragma unroll
    for (int r = 0; r < 8; ++r) {
      P[(mrow + r) * LDSP + r16]      = f32_to_bf16(p0[r]);
      P[(mrow + r) * LDSP + 16 + r16] = f32_to_bf16(p1[r]);
    }
    frag16 fp;
    {
      const unsigned short* pp = &P[r16 * LDSP];
      fp.q[0] = *(const uint4*)(pp + ak);
      fp.q[1] = *(const uint4*)(pp + ak + 8);
    }
    frag16 fv0, fv1;
    {
      const unsigned short* vp0 = &vt[(size_t)(h * 32 + r16) * 4096 + kk];
      fv0.q[0] = *(const uint4*)(vp0 + bk);
      fv0.q[1] = *(const uint4*)(vp0 + bk + 8);
      const unsigned short* vp1 = &vt[(size_t)(h * 32 + 16 + r16) * 4096 + kk];
      fv1.q[0] = *(const uint4*)(vp1 + bk);
      fv1.q[1] = *(const uint4*)(vp1 + bk + 8);
    }
    o0 = __builtin_amdgcn_wmma_f32_16x16x32_bf16(false, fp.v, false, fv0.v, (short)0, o0, false, false);
    o1 = __builtin_amdgcn_wmma_f32_16x16x32_bf16(false, fp.v, false, fv1.v, (short)0, o1, false, false);
  }

#pragma unroll
  for (int r = 0; r < 8; ++r) {
    const int row  = q0 + mrow + r;
    const float il = 1.f / l8[r];
    outH[(size_t)row * 256 + h * 32 + r16]      = f32_to_bf16(o0[r] * il);
    outH[(size_t)row * 256 + h * 32 + 16 + r16] = f32_to_bf16(o1[r] * il);
  }
}

// ---------------------------------------------------------------------------
// small conversion / init kernels
// ---------------------------------------------------------------------------
__global__ void cvt_f32_bf16_k(const float* __restrict__ in,
                               unsigned short* __restrict__ out, int n)
{
  const int i = blockIdx.x * 256 + threadIdx.x;
  if (i < n) out[i] = f32_to_bf16(in[i]);
}

__global__ void prep_inputs_k(const float* __restrict__ cam, const float* __restrict__ lid,
                              unsigned short* __restrict__ camH, unsigned short* __restrict__ lidH,
                              unsigned short* __restrict__ catH)
{
  const int i = blockIdx.x * 256 + threadIdx.x;  // 4096*256
  const int p = i >> 8, c = i & 255;
  const unsigned short a = f32_to_bf16(cam[(size_t)c * 4096 + p]);
  const unsigned short b = f32_to_bf16(lid[(size_t)c * 4096 + p]);
  camH[(size_t)p * 256 + c] = a;
  lidH[(size_t)p * 256 + c] = b;
  catH[(size_t)p * 512 + c] = a;
  catH[(size_t)p * 512 + 256 + c] = b;
}

__global__ void convw_k(const float* __restrict__ w, unsigned short* __restrict__ out)
{
  const int i = blockIdx.x * 256 + threadIdx.x;  // 256*512*9
  const int o = i / (512 * 9);
  const int rem = i - o * 512 * 9;
  const int c = rem / 9;
  const int t = rem - c * 9;
  out[(size_t)o * 4608 + t * 512 + c] = f32_to_bf16(w[i]);
}

__global__ void zero_k(float* __restrict__ p, int n)
{
  const int i = blockIdx.x * 256 + threadIdx.x;
  if (i < n) p[i] = 0.f;
}

// ---------------------------------------------------------------------------
// host orchestration
// ---------------------------------------------------------------------------
extern "C" void kernel_launch(void* const* d_in, const int* in_sizes, int n_in,
                              void* d_out, int out_size, void* d_ws, size_t ws_size,
                              hipStream_t stream)
{
  (void)in_sizes; (void)n_in; (void)out_size; (void)ws_size;

  // pytree leaf order (sorted dict keys): cam_ft, lidar_ft, params{cross[3]{...}, fuser_w, query_embed, self[2]{...}}
  const int IN_CAM = 0, IN_LID = 1;
  auto CB = [](int i) { return 2 + i * 16; };
  const int C_AW_B=0, C_AW_W=1, C_F1B=2, C_F1W=3, C_F2B=4, C_F2W=5,
            C_LN1B=6, C_LN1S=7, C_LN2B=8, C_LN2S=9, C_OFF_B=10, C_OFF_W=11,
            C_OP_B=12, C_OP_W=13, C_VP_B=14, C_VP_W=15;
  const int IN_FUSER = 50, IN_QE = 51;
  auto SB = [](int i) { return 52 + i * 12; };
  const int S_F1B=0, S_F1W=1, S_F2B=2, S_F2W=3, S_INB=4, S_INW=5,
            S_LN1B=6, S_LN1S=7, S_LN2B=8, S_LN2S=9, S_OUTB=10, S_OUTW=11;

  char* ws = (char*)d_ws;
  size_t off = 0;
  auto alloc = [&](size_t bytes) { size_t r = off; off += (bytes + 255) & ~(size_t)255; return ws + r; };

  unsigned short* camH   = (unsigned short*)alloc((size_t)NQ * D * 2);
  unsigned short* lidH   = (unsigned short*)alloc((size_t)NQ * D * 2);
  unsigned short* catH   = (unsigned short*)alloc((size_t)NQ * 512 * 2);
  unsigned short* fusedH = (unsigned short*)alloc((size_t)NQ * D * 2);
  unsigned short* convW  = (unsigned short*)alloc((size_t)256 * 4608 * 2);
  unsigned short* wpool  = (unsigned short*)alloc((size_t)4 * 1024 * 1024);
  float*          tgt    = (float*)alloc((size_t)NQ * D * 4);
  unsigned short* t2H    = (unsigned short*)alloc((size_t)NQ * D * 2);
  unsigned short* t2qH   = (unsigned short*)alloc((size_t)NQ * D * 2);
  float*          valF   = (float*)alloc((size_t)NQ * D * 4);
  float*          offF   = (float*)alloc((size_t)NQ * 128 * 4);
  float*          awF    = (float*)alloc((size_t)NQ * 64 * 4);
  unsigned short* sampH  = (unsigned short*)alloc((size_t)NQ * D * 2);
  unsigned short* ffhH   = (unsigned short*)alloc((size_t)NQ * D * 2);
  unsigned short* qkH    = (unsigned short*)alloc((size_t)NQ * 512 * 2);
  float*          vF     = (float*)alloc((size_t)NQ * D * 4);
  unsigned short* vtH    = (unsigned short*)alloc((size_t)NQ * D * 2);
  unsigned short* attH   = (unsigned short*)alloc((size_t)NQ * D * 2);

  size_t wcur = 0;
  auto cvtW = [&](int idx, int n) -> unsigned short* {
    unsigned short* dst = wpool + wcur;
    wcur += (size_t)((n + 127) & ~127);
    cvt_f32_bf16_k<<<(n + 255) / 256, 256, 0, stream>>>((const float*)d_in[idx], dst, n);
    return dst;
  };

  auto gemm = [&](const unsigned short* A, const unsigned short* B, const float* bias,
                  const float* res, float* outF, unsigned short* outH,
                  int N, int K, int ldo, int relu) {
    const int padcode = (K == 512) ? 7 : 6;  // B row length in DWORDs: 256 -> code7, 128 -> code6
    gemm_bf16_k<<<dim3(NQ / BM, N / BN), 256, 0, stream>>>(
        A, K, B, K, bias, res, outF, outH, ldo, K, relu, padcode);
  };

  const float* qpos = (const float*)d_in[IN_QE];

  // --- input conversion, conv weight reorder, tgt init ---
  prep_inputs_k<<<NQ * D / 256, 256, 0, stream>>>(
      (const float*)d_in[IN_CAM], (const float*)d_in[IN_LID], camH, lidH, catH);
  convw_k<<<(256 * 512 * 9) / 256, 256, 0, stream>>>((const float*)d_in[IN_FUSER], convW);
  zero_k<<<NQ * D / 256, 256, 0, stream>>>(tgt, NQ * D);

  // --- fused = relu(conv3x3(concat)) ---
  conv3x3_wmma_k<<<dim3(NQ / BM, D / BN), 256, 0, stream>>>(catH, convW, fusedH);

  // --- weight bf16 conversions ---
  unsigned short *c_off[3], *c_aw[3], *c_vp[3], *c_op[3], *c_f1[3], *c_f2[3];
  for (int i = 0; i < 3; ++i) {
    const int cb = CB(i);
    c_off[i] = cvtW(cb + C_OFF_W, 128 * 256);
    c_aw[i]  = cvtW(cb + C_AW_W, 64 * 256);
    c_vp[i]  = cvtW(cb + C_VP_W, 256 * 256);
    c_op[i]  = cvtW(cb + C_OP_W, 256 * 256);
    c_f1[i]  = cvtW(cb + C_F1W, 256 * 256);
    c_f2[i]  = cvtW(cb + C_F2W, 256 * 256);
  }
  unsigned short *s_in[2], *s_out[2], *s_f1[2], *s_f2[2];
  for (int i = 0; i < 2; ++i) {
    const int sb = SB(i);
    s_in[i]  = cvtW(sb + S_INW, 768 * 256);
    s_out[i] = cvtW(sb + S_OUTW, 256 * 256);
    s_f1[i]  = cvtW(sb + S_F1W, 256 * 256);
    s_f2[i]  = cvtW(sb + S_F2W, 256 * 256);
  }

  // --- cross blocks: src = lidar, cam, fused ---
  const unsigned short* srcs[3] = { lidH, camH, fusedH };
  for (int i = 0; i < 3; ++i) {
    const int cb = CB(i);
    const float* ln1s = (const float*)d_in[cb + C_LN1S];
    const float* ln1b = (const float*)d_in[cb + C_LN1B];
    const float* ln2s = (const float*)d_in[cb + C_LN2S];
    const float* ln2b = (const float*)d_in[cb + C_LN2B];

    layernorm_k<<<NQ / 8, 256, 0, stream>>>(tgt, ln1s, ln1b, qpos, (unsigned short*)nullptr, t2qH);
    gemm(srcs[i], c_vp[i], (const float*)d_in[cb + C_VP_B], nullptr, valF, nullptr, 256, 256, 256, 0);
    gemm(t2qH, c_off[i], (const float*)d_in[cb + C_OFF_B], nullptr, offF, nullptr, 128, 256, 128, 0);
    gemm(t2qH, c_aw[i], (const float*)d_in[cb + C_AW_B], nullptr, awF, nullptr, 64, 256, 64, 0);
    msda_k<<<NQ, 256, 0, stream>>>(offF, awF, valF, sampH);
    gemm(sampH, c_op[i], (const float*)d_in[cb + C_OP_B], tgt, tgt, nullptr, 256, 256, 256, 0);

    layernorm_k<<<NQ / 8, 256, 0, stream>>>(tgt, ln2s, ln2b, qpos, t2H, (unsigned short*)nullptr);
    gemm(t2H, c_f1[i], (const float*)d_in[cb + C_F1B], nullptr, nullptr, ffhH, 256, 256, 256, 1);
    gemm(ffhH, c_f2[i], (const float*)d_in[cb + C_F2B], tgt, tgt, nullptr, 256, 256, 256, 1);
  }

  // --- self blocks ---
  for (int i = 0; i < 2; ++i) {
    const int sb = SB(i);
    const float* ln1s = (const float*)d_in[sb + S_LN1S];
    const float* ln1b = (const float*)d_in[sb + S_LN1B];
    const float* ln2s = (const float*)d_in[sb + S_LN2S];
    const float* ln2b = (const float*)d_in[sb + S_LN2B];
    const float* inb  = (const float*)d_in[sb + S_INB];

    layernorm_k<<<NQ / 8, 256, 0, stream>>>(tgt, ln1s, ln1b, qpos, t2H, t2qH);
    gemm(t2qH, s_in[i], inb, nullptr, nullptr, qkH, 512, 256, 512, 0);               // Q|K
    gemm(t2H, s_in[i] + (size_t)512 * 256, inb + 512, nullptr, vF, nullptr, 256, 256, 256, 0); // V
    vtrans_k<<<NQ * D / 256, 256, 0, stream>>>(vF, vtH);
    flash_k<<<(HEADS * 256) / 8, 256, 0, stream>>>(qkH, vtH, attH);
    gemm(attH, s_out[i], (const float*)d_in[sb + S_OUTB], tgt, tgt, nullptr, 256, 256, 256, 0);

    layernorm_k<<<NQ / 8, 256, 0, stream>>>(tgt, ln2s, ln2b, qpos, t2H, (unsigned short*)nullptr);
    gemm(t2H, s_f1[i], (const float*)d_in[sb + S_F1B], nullptr, nullptr, ffhH, 256, 256, 256, 1);
    float* fout = (i == 1) ? (float*)d_out : tgt;
    gemm(ffhH, s_f2[i], (const float*)d_in[sb + S_F2B], tgt, fout, nullptr, 256, 256, 256, 1);
  }
}
